// Learned_QCQP_58222576664847
// MI455X (gfx1250) — matrix operand order repaired and checked by reading, not compiled, not run on hardware
//
#include <hip/hip_runtime.h>
#include <hip/hip_bf16.h>
#include <math.h>

// ---------------- problem constants ----------------
#define B_TOT   2048
#define NUMT    100
#define NVAR    11
#define NOBS    10
#define NCIR    3
#define T_FIN   10.0f
#define HDIM    256
#define G3      768          // 3*H gate width

#define RHO_OBS   1.0f
#define RHO_INEQ  1.0f
#define RHO_LANE  1.0f
#define RHO_PROJ  1.0f
#define W_SMOOTH  20.0f
#define W_VDES    2.0f
#define W_YDES    1.0f
#define A_EGO     3.0f
#define B_EGO     1.5f
#define MARGIN    0.2f
#define V_MIN     0.001f
#define V_MAX     18.0f
#define A_MAX     6.0f
#define MAXITER   5

// ---------------- WMMA vector types (gfx1250, wave32) ----------------
typedef _Float16 h16;
typedef __attribute__((ext_vector_type(16))) _Float16 v16h;
typedef __attribute__((ext_vector_type(8)))  _Float16 v8h;
typedef __attribute__((ext_vector_type(8)))  float    v8f;

// ---------------- workspace layout ----------------
// float region (offsets in floats)
#define WS_P      0        // [100*11]
#define WS_PD     1100
#define WS_PDD    2200
#define WS_SUMP   3300     // [11]
#define WS_SUMPD  3311     // [11]
#define WS_KX     3328     // 15x15 inverse
#define WS_KY     3584     // 14x14 inverse
#define WS_KIX    3840     // 15x15 inverse
#define WS_KIY    4096     // 14x14 inverse
#define WS_LAM    4352     // [2048*22]
// half region starts at float offset 49408 (byte 197632)
#define WS_HALF_BYTE 197632
#define WHH_PK_OFF 0        // 8*48*32*16   = 196608 halves
#define W1_PK_OFF  196608   // 8*32*32*16   = 131072 halves
#define W2_PK_OFF  327680   // 16*2*32*16   = 16384  halves
#define PK_TOTAL   344064
// total ws use: 197632 + 2*344064 = 885,760 bytes

// ---------------- prep: bases + KKT inverses ----------------
__device__ void gj_invert(float* A, float* Inv, int n) {
    for (int i = 0; i < n; i++)
        for (int j = 0; j < n; j++) Inv[i*n+j] = (i == j) ? 1.f : 0.f;
    for (int c = 0; c < n; c++) {
        int p = c; float mx = fabsf(A[c*n+c]);
        for (int r = c+1; r < n; r++) { float v = fabsf(A[r*n+c]); if (v > mx) { mx = v; p = r; } }
        if (p != c) {
            for (int j = 0; j < n; j++) {
                float t = A[c*n+j]; A[c*n+j] = A[p*n+j]; A[p*n+j] = t;
                t = Inv[c*n+j]; Inv[c*n+j] = Inv[p*n+j]; Inv[p*n+j] = t;
            }
        }
        float inv = 1.f / A[c*n+c];
        for (int j = 0; j < n; j++) { A[c*n+j] *= inv; Inv[c*n+j] *= inv; }
        for (int r = 0; r < n; r++) {
            if (r == c) continue;
            float f = A[r*n+c];
            for (int j = 0; j < n; j++) { A[r*n+j] -= f*A[c*n+j]; Inv[r*n+j] -= f*Inv[c*n+j]; }
        }
    }
}

__global__ __launch_bounds__(256) void prep_kernel(float* __restrict__ wsf) {
    __shared__ float Ps[1100], Pds[1100], Pdds[1100];
    __shared__ float PTPs[121], PdTPds[121], PddTPdds[121];
    __shared__ float Kbuf[225], Ibuf[225];
    const int tid = threadIdx.x;

    if (tid < NUMT) {
        float s = (float)tid / (float)(NUMT - 1);
        float pw[NVAR];
        pw[0] = 1.f;
        for (int j = 1; j < NVAR; j++) pw[j] = pw[j-1] * s;
        for (int j = 0; j < NVAR; j++) {
            Ps  [tid*NVAR+j] = pw[j];
            Pds [tid*NVAR+j] = (j >= 1) ? (float)j * pw[j-1] / T_FIN : 0.f;
            Pdds[tid*NVAR+j] = (j >= 2) ? (float)(j*(j-1)) * pw[j-2] / (T_FIN*T_FIN) : 0.f;
        }
    }
    __syncthreads();

    if (tid < 121) {
        int a = tid / NVAR, c = tid % NVAR;
        float s0 = 0, s1 = 0, s2 = 0;
        for (int t = 0; t < NUMT; t++) {
            s0 += Ps  [t*NVAR+a] * Ps  [t*NVAR+c];
            s1 += Pds [t*NVAR+a] * Pds [t*NVAR+c];
            s2 += Pdds[t*NVAR+a] * Pdds[t*NVAR+c];
        }
        PTPs[tid] = s0; PdTPds[tid] = s1; PddTPdds[tid] = s2;
    }
    for (int i = tid; i < 1100; i += 256) {
        wsf[WS_P+i] = Ps[i]; wsf[WS_PD+i] = Pds[i]; wsf[WS_PDD+i] = Pdds[i];
    }
    if (tid < NVAR) {
        float sA = 0, sB = 0;
        for (int t = 0; t < NUMT; t++) { sA += Ps[t*NVAR+tid]; sB += Pds[t*NVAR+tid]; }
        wsf[WS_SUMP+tid] = sA; wsf[WS_SUMPD+tid] = sB;
    }
    __syncthreads();

    if (tid == 0) {
        for (int sys = 0; sys < 4; sys++) {
            const int meq = (sys == 0 || sys == 2) ? 4 : 3;
            const int n = NVAR + meq;
            for (int i = 0; i < n*n; i++) Kbuf[i] = 0.f;
            for (int i = 0; i < NVAR; i++) {
                for (int j = 0; j < NVAR; j++) {
                    float q;
                    if (sys <= 1) {
                        q = ((i == j) ? (RHO_PROJ + W_SMOOTH) : 0.f)
                          + RHO_OBS * (float)(NOBS*NCIR) * PTPs[i*NVAR+j]
                          + RHO_INEQ * (PdTPds[i*NVAR+j] + PddTPdds[i*NVAR+j]);
                        if (sys == 1) q += 2.f * RHO_LANE * PTPs[i*NVAR+j];
                    } else {
                        q = ((i == j) ? W_SMOOTH : 0.f) + W_VDES * PdTPds[i*NVAR+j];
                        if (sys == 3) q += W_YDES * PTPs[i*NVAR+j];
                    }
                    Kbuf[i*n+j] = q;
                }
            }
            for (int r = 0; r < meq; r++) {
                for (int j = 0; j < NVAR; j++) {
                    float a;
                    if      (r == 0) a = Ps  [0*NVAR+j];
                    else if (r == 1) a = Pds [0*NVAR+j];
                    else if (r == 2) a = Pdds[0*NVAR+j];
                    else             a = Ps  [(NUMT-1)*NVAR+j];
                    Kbuf[(NVAR+r)*n + j] = a;
                    Kbuf[j*n + NVAR + r] = a;
                }
            }
            gj_invert(Kbuf, Ibuf, n);
            float* dst = (sys == 0) ? wsf + WS_KX : (sys == 1) ? wsf + WS_KY
                       : (sys == 2) ? wsf + WS_KIX : wsf + WS_KIY;
            for (int i = 0; i < n*n; i++) dst[i] = Ibuf[i];
        }
    }
}

// ---------------- weight packing to WMMA B-matrix lane layout ----------------
// B fragment (32x16 f16, K x N): packed[((kt*NT + nt)*32 + lane)*16 + e]
//   = W[kt*32 + (lane>=16 ? 8 : 0) + (e&7) + (e>=8 ? 16 : 0)][nt*16 + (lane & 15)]
__global__ __launch_bounds__(256) void pack_weights(
    const float* __restrict__ Whh, const float* __restrict__ W1, const float* __restrict__ W2,
    h16* __restrict__ Whh_pk, h16* __restrict__ W1_pk, h16* __restrict__ W2_pk)
{
    int idx = blockIdx.x * blockDim.x + threadIdx.x;
    if (idx >= PK_TOTAL) return;
    if (idx < W1_PK_OFF) {                         // W_hh: 256x768, 8 ktiles x 48 ntiles
        int e = idx & 15, lane = (idx >> 4) & 31;
        int r = idx >> 9;
        int nt = r % 48, kt = r / 48;
        int k = kt*32 + ((lane >> 4) << 3) + (e & 7) + ((e >> 3) << 4);
        int n = nt*16 + (lane & 15);
        Whh_pk[idx] = (h16)Whh[k*G3 + n];
    } else if (idx < W2_PK_OFF) {                  // W1: 256x512, 8 x 32
        int li = idx - W1_PK_OFF;
        int e = li & 15, lane = (li >> 4) & 31;
        int r = li >> 9;
        int nt = r & 31, kt = r >> 5;
        int k = kt*32 + ((lane >> 4) << 3) + (e & 7) + ((e >> 3) << 4);
        int n = nt*16 + (lane & 15);
        W1_pk[li] = (h16)W1[k*512 + n];
    } else {                                       // W2: 512x22 padded to 512x32, 16 x 2
        int li = idx - W2_PK_OFF;
        int e = li & 15, lane = (li >> 4) & 31;
        int r = li >> 9;
        int nt = r & 1, kt = r >> 1;
        int k = kt*32 + ((lane >> 4) << 3) + (e & 7) + ((e >> 3) << 4);
        int n = nt*16 + (lane & 15);
        W2_pk[li] = (n < 22) ? (h16)W2[k*22 + n] : (h16)0.0f;
    }
}

// ---------------- fragment loaders ----------------
// A fragment (16x32 f16, M x K), per ISA layout: lane L<16 holds row M=L with
// elements 0..7 = K k0..k0+7, 8..15 = K k0+16..k0+23; lanes 16..31 shifted by +8.
__device__ __forceinline__ v16h load_afrag(const h16* rowp) {
    v8h lo = *(const v8h*)rowp;          // 8 halves
    v8h hi = *(const v8h*)(rowp + 16);   // skip 16 halves forward
    v16h a;
#pragma unroll
    for (int e = 0; e < 8; e++) { a[e] = lo[e]; a[e+8] = hi[e]; }
    return a;
}
__device__ __forceinline__ v16h load_bfrag(const h16* p) {
    v8h lo = *(const v8h*)p;
    v8h hi = *(const v8h*)(p + 8);
    v16h b;
#pragma unroll
    for (int e = 0; e < 8; e++) { b[e] = lo[e]; b[e+8] = hi[e]; }
    return b;
}

// ---------------- GRU + MLP (WMMA) ----------------
#define MT    32          // batch rows per block
#define HPAD  264         // padded f16 h row (256+8), 528B = 33*16B
#define GPAD  776         // padded f32 gate row
#define APAD  520         // padded f16 act row (512+8), 1040B = 65*16B
#define SMEM_GRU (MT*HPAD*2 + MT*GPAD*4 + MT*APAD*2 + 5*G3*4 + G3*4)

__global__ __launch_bounds__(256) void gru_mlp_kernel(
    const float* __restrict__ cobs, const float* __restrict__ psi,
    const float* __restrict__ dimx, const float* __restrict__ dimy,
    const float* __restrict__ Wih,  const float* __restrict__ gbias,
    const float* __restrict__ h0,   const float* __restrict__ b1,
    const float* __restrict__ b2,
    const h16* __restrict__ Whh_pk, const h16* __restrict__ W1_pk,
    const h16* __restrict__ W2_pk,  float* __restrict__ lam_out)
{
    extern __shared__ __align__(16) unsigned char smem[];
    h16*   hbuf   = (h16*)  smem;                                       // [MT][HPAD]
    float* ghbuf  = (float*)(smem + MT*HPAD*2);                         // [MT][GPAD]
    h16*   actbuf = (h16*)  (smem + MT*HPAD*2 + MT*GPAD*4);             // [MT][APAD]
    float* wih_s  = (float*)(smem + MT*HPAD*2 + MT*GPAD*4 + MT*APAD*2); // [5*768]
    float* gb_s   = wih_s + 5*G3;                                       // [768]

    const int tid   = threadIdx.x;
    const int lane  = tid & 31;
    const int wid   = tid >> 5;            // 8 waves
    const int bBase = blockIdx.x * MT;

    const int wm = wid >> 2;               // 0..1 : 16-row half
    const int wn = wid & 3;                // 0..3 : column group
    const int mrow  = wm*16 + (lane & 15); // A-matrix row for this lane
    const int khalf = lane >> 4;           // 0/1 : K-half select

    // stage W_ih, bias, init h
    for (int i = tid; i < 5*G3; i += 256) wih_s[i] = Wih[i];
    for (int i = tid; i < G3;   i += 256) gb_s[i]  = gbias[i];
    for (int idx = tid; idx < MT*HDIM; idx += 256) {
        int m = idx >> 8, k = idx & 255;
        hbuf[m*HPAD + k] = (h16)h0[k];
    }
    __syncthreads();

    // ---- GRU recurrence: gh = h @ W_hh via WMMA, gates in VALU ----
    // Keep the B-fragment loads INSIDE the timestep loop (weights are
    // L2-resident; hoisting them across 10 steps spills to scratch).
    // The opaque ZERO OFFSET defeats LICM while keeping the pointer's
    // global address space inferable (global_load, not flat_load).
    unsigned int whh_off = 0u;
#pragma unroll 1
    for (int step = 0; step < NOBS; ++step) {
        asm volatile("" : "+s"(whh_off));              // opaque 0, redefined per step
        const h16* whh_base = Whh_pk + whh_off;

        v8f acc[12];
#pragma unroll
        for (int nt = 0; nt < 12; ++nt) acc[nt] = (v8f){};
#pragma unroll
        for (int kt = 0; kt < 8; ++kt) {
            const h16* ap = &hbuf[mrow*HPAD + kt*32 + 8*khalf];
            v16h afrag = load_afrag(ap);
#pragma unroll
            for (int nt = 0; nt < 12; ++nt) {
                int ntg = wn*12 + nt;
                v16h bfrag = load_bfrag(&whh_base[(((size_t)kt*48 + ntg)*32 + lane)*16]);
                acc[nt] = __builtin_amdgcn_wmma_f32_16x16x32_f16(
                    false, afrag, false, bfrag, (short)0, acc[nt], false, false);
            }
        }
        // C layout: VGPR r -> row wm*16 + 8*khalf + r, col = tile + (lane&15)
        {
            int mb = wm*16 + 8*khalf;
#pragma unroll
            for (int nt = 0; nt < 12; ++nt) {
                int ncol = wn*192 + nt*16 + (lane & 15);
#pragma unroll
                for (int r = 0; r < 8; ++r)
                    ghbuf[(mb + r)*GPAD + ncol] = acc[nt][r];
            }
        }
        __syncthreads();

        // gates + state update
        for (int idx = tid; idx < MT*HDIM; idx += 256) {
            int m = idx >> 8, j = idx & 255;
            int bg = bBase + m;
            float xo  = cobs[bg*20 + step];
            float yo  = cobs[bg*20 + 10 + step];
            float dx  = dimx[bg*10 + step];
            float dyv = dimy[bg*10 + step];
            float ps  = psi [bg*10 + step];
            float gxz = xo*wih_s[0*G3+j]     + yo*wih_s[1*G3+j]     + dx*wih_s[2*G3+j]
                      + dyv*wih_s[3*G3+j]    + ps*wih_s[4*G3+j]     + gb_s[j];
            float gxr = xo*wih_s[0*G3+256+j] + yo*wih_s[1*G3+256+j] + dx*wih_s[2*G3+256+j]
                      + dyv*wih_s[3*G3+256+j]+ ps*wih_s[4*G3+256+j] + gb_s[256+j];
            float gxn = xo*wih_s[0*G3+512+j] + yo*wih_s[1*G3+512+j] + dx*wih_s[2*G3+512+j]
                      + dyv*wih_s[3*G3+512+j]+ ps*wih_s[4*G3+512+j] + gb_s[512+j];
            float ghz = ghbuf[m*GPAD + j];
            float ghr = ghbuf[m*GPAD + 256 + j];
            float ghn = ghbuf[m*GPAD + 512 + j];
            float z = 1.f / (1.f + __expf(-(gxz + ghz)));
            float r = 1.f / (1.f + __expf(-(gxr + ghr)));
            float n = tanhf(gxn + r*ghn);
            float hp = (float)hbuf[m*HPAD + j];
            hbuf[m*HPAD + j] = (h16)((1.f - z)*n + z*hp);
        }
        __syncthreads();
    }

    // ---- MLP layer 1: act = relu(hT @ W1 + b1)  (32 x 512) ----
    {
        v8f acc2[8];
#pragma unroll
        for (int nt = 0; nt < 8; ++nt) acc2[nt] = (v8f){};
#pragma unroll
        for (int kt = 0; kt < 8; ++kt) {
            v16h afrag = load_afrag(&hbuf[mrow*HPAD + kt*32 + 8*khalf]);
#pragma unroll
            for (int nt = 0; nt < 8; ++nt) {
                int ntg = wn*8 + nt;
                v16h bfrag = load_bfrag(&W1_pk[(((size_t)kt*32 + ntg)*32 + lane)*16]);
                acc2[nt] = __builtin_amdgcn_wmma_f32_16x16x32_f16(
                    false, afrag, false, bfrag, (short)0, acc2[nt], false, false);
            }
        }
        int mb = wm*16 + 8*khalf;
#pragma unroll
        for (int nt = 0; nt < 8; ++nt) {
            int ncol = wn*128 + nt*16 + (lane & 15);
            float bias = b1[ncol];
#pragma unroll
            for (int r = 0; r < 8; ++r)
                actbuf[(mb + r)*APAD + ncol] = (h16)fmaxf(acc2[nt][r] + bias, 0.f);
        }
    }
    __syncthreads();

    // ---- MLP layer 2: lam = act @ W2 + b2  (32 x 22, padded to 32) ----
    if (wid < 4) {
        int wm2 = wid >> 1, wn2 = wid & 1;
        int mrow2 = wm2*16 + (lane & 15);
        v8f acc3 = (v8f){};
#pragma unroll
        for (int kt = 0; kt < 16; ++kt) {
            v16h afrag = load_afrag(&actbuf[mrow2*APAD + kt*32 + 8*khalf]);
            v16h bfrag = load_bfrag(&W2_pk[(((size_t)kt*2 + wn2)*32 + lane)*16]);
            acc3 = __builtin_amdgcn_wmma_f32_16x16x32_f16(
                false, afrag, false, bfrag, (short)0, acc3, false, false);
        }
        int ncol = wn2*16 + (lane & 15);
        int mb = wm2*16 + 8*khalf;
        if (ncol < 22) {
#pragma unroll
            for (int r = 0; r < 8; ++r)
                lam_out[(size_t)(bBase + mb + r)*22 + ncol] = acc3[r] + b2[ncol];
        }
    }
}

// ---------------- ADMM solver: one block per batch row ----------------
__global__ __launch_bounds__(128) void solver_kernel(
    const float* __restrict__ init, const float* __restrict__ goal,
    const float* __restrict__ cobs, const float* __restrict__ psi,
    const float* __restrict__ dimx, const float* __restrict__ dimy,
    const float* __restrict__ yub,  const float* __restrict__ ylb,
    const float* __restrict__ wsf,  float* __restrict__ out)
{
    __shared__ float Ps[1100], Pds[1100], Pdds[1100];
    __shared__ float xs[100], xds[100], xdds[100], ys[100], yds[100], ydds[100];
    __shared__ float sbox[100], sboy[100], bvx[100], bvy[100], bax[100], bay[100];
    __shared__ float blA[100], blB[100];
    __shared__ float xob[30], yob[30], aob[30], bob[30];
    __shared__ float cx[11], cy[11], lx[11], ly[11], cx0[11], cy0[11], topx[11], topy[11];
    __shared__ float beqx[4], beqy[3], scal[4];   // vdes, goal_y, y_ub, y_lb

    const int b = blockIdx.x, tid = threadIdx.x;
    for (int i = tid; i < 1100; i += 128) {
        Ps[i] = wsf[WS_P+i]; Pds[i] = wsf[WS_PD+i]; Pdds[i] = wsf[WS_PDD+i];
    }
    if (tid < NOBS*NCIR) {
        int i = tid / NCIR, k = tid % NCIR;
        float dx = dimx[b*10+i], dyv = dimy[b*10+i], ps = psi[b*10+i];
        float r = (k == 0) ? 0.f : ((k == 1) ? 0.5f*(dx - dyv) : -0.5f*(dx - dyv));
        float sp, cp; sincosf(ps, &sp, &cp);
        xob[tid] = cobs[b*20 + i]      + r*cp;
        yob[tid] = cobs[b*20 + 10 + i] + r*sp;
        float rad = 0.5f*dyv;
        aob[tid] = A_EGO + rad + MARGIN;
        bob[tid] = B_EGO + rad + MARGIN;
    }
    if (tid == 0) {
        float x0 = init[b*6], y0 = init[b*6+1], vx0 = init[b*6+2];
        float vy0 = init[b*6+3], ax0 = init[b*6+4], ay0 = init[b*6+5];
        beqx[0] = x0; beqx[1] = vx0; beqx[2] = ax0; beqx[3] = goal[b*2];
        beqy[0] = y0; beqy[1] = vy0; beqy[2] = ay0;
        scal[0] = sqrtf(vx0*vx0 + vy0*vy0);
        scal[1] = goal[b*2+1];
        scal[2] = yub[b]; scal[3] = ylb[b];
    }
    __syncthreads();

    if (tid < NVAR) {
        const float* KIX = wsf + WS_KIX;
        const float* KIY = wsf + WS_KIY;
        float sx = 0.f, sy = 0.f;
        for (int j = 0; j < NVAR; j++) {
            sx += KIX[tid*15+j] * (W_VDES * scal[0] * wsf[WS_SUMPD+j]);
            sy += KIY[tid*14+j] * (W_YDES * scal[1] * wsf[WS_SUMP+j]);
        }
        for (int j = 0; j < 4; j++) sx += KIX[tid*15+11+j] * beqx[j];
        for (int j = 0; j < 3; j++) sy += KIY[tid*14+11+j] * beqy[j];
        cx0[tid] = sx; cy0[tid] = sy; cx[tid] = sx; cy[tid] = sy;
        const float* lamp = wsf + WS_LAM + (size_t)b*22;
        lx[tid] = lamp[tid]; ly[tid] = lamp[11+tid];
    }
    __syncthreads();

    for (int it = 0; it < MAXITER; ++it) {
        // trajectories from current coefficients
        if (tid < NUMT) {
            float a0=0,a1=0,a2=0,c0=0,c1=0,c2=0;
            for (int j = 0; j < NVAR; j++) {
                float pc = Ps[tid*11+j], pd = Pds[tid*11+j], pdd = Pdds[tid*11+j];
                a0 += cx[j]*pc; a1 += cx[j]*pd; a2 += cx[j]*pdd;
                c0 += cy[j]*pc; c1 += cy[j]*pd; c2 += cy[j]*pdd;
            }
            xs[tid]=a0; xds[tid]=a1; xdds[tid]=a2; ys[tid]=c0; yds[tid]=c1; ydds[tid]=c2;
        }
        __syncthreads();
        // per-time-step projections (obstacles / velocity / accel / lane)
        if (tid < NUMT) {
            float xt = xs[tid], yt = ys[tid];
            float sx = 0.f, sy = 0.f;
            for (int o = 0; o < NOBS*NCIR; o++) {
                float wc = xt - xob[o], wv = yt - yob[o];
                float a = aob[o], bb = bob[o];
                float al = atan2f(wv*a, wc*bb);
                float sa, ca; sincosf(al, &sa, &ca);
                float c1o = RHO_OBS*(a*a*ca*ca + bb*bb*sa*sa);
                float c2o = RHO_OBS*(a*wc*ca + bb*wv*sa);
                float d = fmaxf(1.f, c2o/c1o);
                sx += xob[o] + a*d*ca;
                sy += yob[o] + bb*d*sa;
            }
            sbox[tid] = sx; sboy[tid] = sy;
            float xd = xds[tid], yd = yds[tid];
            float av = fminf(0.41f, fmaxf(-0.41f, atan2f(yd, xd)));
            float sv, cv; sincosf(av, &sv, &cv);
            float dv = fminf(V_MAX, fmaxf(V_MIN, xd*cv + yd*sv));
            bvx[tid] = dv*cv; bvy[tid] = dv*sv;
            float xdd = xdds[tid], ydd = ydds[tid];
            float aa = atan2f(ydd + 1e-4f, xdd + 1e-4f);
            float saa, caa; sincosf(aa, &saa, &caa);
            float da = fminf(A_MAX, fmaxf(0.f, xdd*caa + ydd*saa));
            bax[tid] = da*caa; bay[tid] = da*saa;
            blA[tid] = fminf(yt, scal[2]);
            blB[tid] = fminf(-yt, -scal[3]);
        }
        __syncthreads();
        // top_x / top_y
        if (tid < 11) {
            int j = tid; float sA=0,sB=0,sC=0;
            for (int t = 0; t < NUMT; t++) {
                sA += sbox[t]*Ps[t*11+j]; sB += bvx[t]*Pds[t*11+j]; sC += bax[t]*Pdds[t*11+j];
            }
            topx[j] = lx[j] + RHO_PROJ*cx0[j] + RHO_OBS*sA + RHO_INEQ*(sB + sC);
        } else if (tid >= 16 && tid < 27) {
            int j = tid - 16; float sA=0,sB=0,sC=0,sD=0;
            for (int t = 0; t < NUMT; t++) {
                sA += sboy[t]*Ps[t*11+j]; sB += bvy[t]*Pds[t*11+j];
                sC += bay[t]*Pdds[t*11+j]; sD += (blA[t]-blB[t])*Ps[t*11+j];
            }
            topy[j] = ly[j] + RHO_PROJ*cy0[j] + RHO_OBS*sA + RHO_INEQ*(sB + sC) + RHO_LANE*sD;
        }
        __syncthreads();
        // KKT solves (precomputed inverses)
        if (tid < 11) {
            const float* KX = wsf + WS_KX; float s = 0.f;
            for (int j = 0; j < 11; j++) s += KX[tid*15+j]*topx[j];
            for (int j = 0; j < 4;  j++) s += KX[tid*15+11+j]*beqx[j];
            cx[tid] = s;
        } else if (tid >= 16 && tid < 27) {
            int i = tid - 16; const float* KY = wsf + WS_KY; float s = 0.f;
            for (int j = 0; j < 11; j++) s += KY[i*14+j]*topy[j];
            for (int j = 0; j < 3;  j++) s += KY[i*14+11+j]*beqy[j];
            cy[i] = s;
        }
        __syncthreads();
        // new trajectories
        if (tid < NUMT) {
            float a0=0,a1=0,a2=0,c0=0,c1=0,c2=0;
            for (int j = 0; j < NVAR; j++) {
                float pc = Ps[tid*11+j], pd = Pds[tid*11+j], pdd = Pdds[tid*11+j];
                a0 += cx[j]*pc; a1 += cx[j]*pd; a2 += cx[j]*pdd;
                c0 += cy[j]*pc; c1 += cy[j]*pd; c2 += cy[j]*pdd;
            }
            xs[tid]=a0; xds[tid]=a1; xdds[tid]=a2; ys[tid]=c0; yds[tid]=c1; ydds[tid]=c2;
        }
        __syncthreads();
        // dual updates
        if (tid < 11) {
            int j = tid; float s1=0,s2=0,s3=0;
            for (int t = 0; t < NUMT; t++) {
                s1 += (30.f*xs[t] - sbox[t])*Ps[t*11+j];
                s2 += (xds[t]  - bvx[t])*Pds[t*11+j];
                s3 += (xdds[t] - bax[t])*Pdds[t*11+j];
            }
            lx[j] -= RHO_OBS*s1 + RHO_INEQ*(s2 + s3);
        } else if (tid >= 16 && tid < 27) {
            int j = tid - 16; float s1=0,s2=0,s3=0,s4=0;
            for (int t = 0; t < NUMT; t++) {
                s1 += (30.f*ys[t] - sboy[t])*Ps[t*11+j];
                s2 += (yds[t]  - bvy[t])*Pds[t*11+j];
                s3 += (ydds[t] - bay[t])*Pdds[t*11+j];
                s4 += (2.f*ys[t] - blA[t] + blB[t])*Ps[t*11+j];
            }
            ly[j] -= RHO_OBS*s1 + RHO_INEQ*(s2 + s3) + RHO_LANE*s4;
        }
        __syncthreads();
    }

    if (tid < 11) {
        out[(size_t)b*22 + tid]      = cx[tid];
        out[(size_t)b*22 + 11 + tid] = cy[tid];
    }
}

// ---------------- launcher ----------------
extern "C" void kernel_launch(void* const* d_in, const int* in_sizes, int n_in,
                              void* d_out, int out_size, void* d_ws, size_t ws_size,
                              hipStream_t stream)
{
    const float* init = (const float*)d_in[0];
    const float* goal = (const float*)d_in[1];
    const float* cobs = (const float*)d_in[2];
    const float* psi  = (const float*)d_in[3];
    const float* dimx = (const float*)d_in[4];
    const float* dimy = (const float*)d_in[5];
    const float* yub  = (const float*)d_in[6];
    const float* ylb  = (const float*)d_in[7];
    const float* Wih  = (const float*)d_in[8];
    const float* Whh  = (const float*)d_in[9];
    const float* gb   = (const float*)d_in[10];
    const float* h0   = (const float*)d_in[11];
    const float* W1   = (const float*)d_in[12];
    const float* b1   = (const float*)d_in[13];
    const float* W2   = (const float*)d_in[14];
    const float* b2   = (const float*)d_in[15];

    float* wsf   = (float*)d_ws;
    h16* hreg    = (h16*)((char*)d_ws + WS_HALF_BYTE);
    h16* Whh_pk  = hreg + WHH_PK_OFF;
    h16* W1_pk   = hreg + W1_PK_OFF;
    h16* W2_pk   = hreg + W2_PK_OFF;
    float* lam   = wsf + WS_LAM;
    float* out   = (float*)d_out;

    prep_kernel<<<1, 256, 0, stream>>>(wsf);
    pack_weights<<<(PK_TOTAL + 255)/256, 256, 0, stream>>>(Whh, W1, W2, Whh_pk, W1_pk, W2_pk);
    gru_mlp_kernel<<<B_TOT/MT, 256, SMEM_GRU, stream>>>(
        cobs, psi, dimx, dimy, Wih, gb, h0, b1, b2, Whh_pk, W1_pk, W2_pk, lam);
    solver_kernel<<<B_TOT, 128, 0, stream>>>(
        init, goal, cobs, psi, dimx, dimy, yub, ylb, wsf, out);
    (void)in_sizes; (void)n_in; (void)out_size; (void)ws_size;
}
